// FlowGuidedDeformAttnAlignV4_6579889897618
// MI455X (gfx1250) — matrix operand order repaired and checked by reading, not compiled, not run on hardware
//
#include <hip/hip_runtime.h>
#include <hip/hip_bf16.h>

// ---- problem constants (match reference) ----
#define BB     4
#define CC     128
#define HH     128
#define WW     128
#define QQ     (HH * WW)      // 16384
#define NH     8
#define NP     8
#define DHD    16             // CC / NH
#define MAXRES 10.0f

typedef _Float16 half_t;
typedef __attribute__((ext_vector_type(16))) _Float16 v16h;
typedef __attribute__((ext_vector_type(8)))  _Float16 v8h;
typedef __attribute__((ext_vector_type(8)))  float    v8f;

// f16 LDS A-tile row stride: 136 halves = 272B = 68 dwords -> bank (4m+k/2)%64,
// conflict-free across the 16 rows, and 16B aligned for ds_load_b128.
#define LDAH 136

// Packed-B geometry: 28 tiles x 4 kc x 32 lanes x 16 halves = 57344 halves.
// Tile order: 0..7 W_val | 8..15 W_off | 16..19 W_attn | 20..27 W_out.
#define NTILES     28
#define PACKB_HALVES (NTILES * 4 * 32 * 16)
#define WS_FLOAT_OFF 131072   // bytes; packed B occupies 114688 B

// Branch-free tanh: tanh(z) = sign(z) * (1 - e^{-2|z|}) / (1 + e^{-2|z|}).
// Argument of exp is <= 0 -> no overflow, no range-split branches (keeps the
// WMMA-carrying waves free of exec-mask churn).
__device__ inline float fast_tanhf(float z) {
    const float t = __expf(-2.0f * fabsf(z));
    return copysignf((1.0f - t) / (1.0f + t), z);
}

// ---------------------------------------------------------------------------
// Fragment loaders (CDNA5 wave32 layouts, CDNA5 ISA §7.12.2)
// ---------------------------------------------------------------------------

// A 16x32 f16 from f16 LDS tile: lane L holds row m=L&15;
// halves i=0..7 at k0+8*(L>>4)+i, i=8..15 at +16. Two 16B ds loads.
__device__ inline v16h load_a_frag(const half_t* lds, int k0, int lane) {
    const half_t* row = lds + (lane & 15) * LDAH + k0 + ((lane >> 4) << 3);
    v8h lo = *(const v8h*)(row);
    v8h hi = *(const v8h*)(row + 16);
    v16h a;
#pragma unroll
    for (int i = 0; i < 8; ++i) { a[i] = lo[i]; a[i + 8] = hi[i]; }
    return a;
}

// B 32x16 f16, pre-packed in fragment order: one aligned 32B load.
__device__ inline v16h load_b_packed(const half_t* wsB, int T, int kc, int lane) {
    return *(const v16h*)(wsB + ((((size_t)T * 4 + kc) * 32) + lane) * 16);
}

// ---------------------------------------------------------------------------
// Kernel 0: pack all weight matrices into f16 WMMA B-fragment order.
// idx = ((T*4 + kc)*32 + lane)*16 + i ; k = kc*32 + 16*(i>>3) + 8*(lane>>4) + (i&7) ;
// col = n0 + (lane&15) ; value = W[k*N + col].
// ---------------------------------------------------------------------------
__global__ __launch_bounds__(256)
void k_prep(const float* __restrict__ W_val, const float* __restrict__ W_off,
            const float* __restrict__ W_attn, const float* __restrict__ W_out,
            half_t* __restrict__ wsB)
{
    const int idx  = blockIdx.x * 256 + threadIdx.x;   // < PACKB_HALVES
    const int i    = idx & 15;
    const int lane = (idx >> 4) & 31;
    const int kc   = (idx >> 9) & 3;
    const int T    = idx >> 11;

    const float* Wm; int N, n0;
    if (T < 8)       { Wm = W_val;  N = 128; n0 = T * 16;        }
    else if (T < 16) { Wm = W_off;  N = 128; n0 = (T - 8) * 16;  }
    else if (T < 20) { Wm = W_attn; N = 64;  n0 = (T - 16) * 16; }
    else             { Wm = W_out;  N = 128; n0 = (T - 20) * 16; }

    const int col = n0 + (lane & 15);
    const int k   = kc * 32 + ((i >> 3) << 4) + ((lane >> 4) << 3) + (i & 7);
    wsB[idx] = (half_t)Wm[(size_t)k * N + col];
}

// ---------------------------------------------------------------------------
// Kernel 1: fused projections (val / sampling locations / attention softmax).
// One 256-thread workgroup (8 wave32) per 16-query strip per batch.
// ---------------------------------------------------------------------------
__global__ __launch_bounds__(256)
void k_proj(const float* __restrict__ nbr, const float* __restrict__ ext,
            const float* __restrict__ flow,
            const float* __restrict__ b_off, const float* __restrict__ b_attn,
            const float* __restrict__ b_val,
            const half_t* __restrict__ wsB,
            float* __restrict__ val_out, float* __restrict__ loc_out,
            float* __restrict__ attn_out)
{
    __shared__ half_t ldsNbr[16 * LDAH];
    __shared__ half_t ldsExt[16 * LDAH];
    __shared__ float  ldsAttn[16 * 64];

    const int wg  = blockIdx.x;            // B * Q/16 workgroups
    const int b   = wg / (QQ / 16);
    const int q0  = (wg % (QQ / 16)) * 16;
    const int tid = threadIdx.x;

    // Stage transposed A tiles as f16 (convert once here, not per use).
    for (int j = tid; j < 2048; j += 256) {
        const int c = j >> 4, m = j & 15;
        const size_t src = (size_t)b * CC * QQ + (size_t)c * QQ + q0 + m;
        ldsNbr[m * LDAH + c] = (half_t)nbr[src];
        ldsExt[m * LDAH + c] = (half_t)ext[src];
    }
    __syncthreads();

    const int lane = tid & 31;
    const int wave = tid >> 5;

    // 20 N-tiles: 0..7 val | 8..15 offsets | 16..19 attn (packed-B tile T == t).
    for (int t = wave; t < 20; t += 8) {   // wave-uniform trip count
        const half_t* A = (t < 8) ? ldsNbr : ldsExt;
        const int n0 = (t < 8) ? t * 16 : (t < 16 ? (t - 8) * 16 : (t - 16) * 16);

        v8f acc = {};
#pragma unroll
        for (int kc = 0; kc < 4; ++kc) {
            v16h af = load_a_frag(A, kc * 32, lane);
            v16h bf = load_b_packed(wsB, t, kc, lane);
            acc = __builtin_amdgcn_wmma_f32_16x16x32_f16(
                false, af, false, bf, (short)0, acc, false, false);
        }

        const int ncol  = n0 + (lane & 15);
        const int mbase = (lane >> 4) * 8;

        if (t < 8) {                                   // value projection
            const float bias = b_val[ncol];
            const int head = ncol >> 4, d = ncol & 15;
#pragma unroll
            for (int r = 0; r < 8; ++r) {
                const int q = q0 + mbase + r;
                val_out[(((size_t)b * NH + head) * QQ + q) * DHD + d] = acc[r] + bias;
            }
        } else if (t < 16) {                           // sampling locations
            const float bias = b_off[ncol];
            const int hp = ncol >> 1;                  // head*P + p
            const int xy = ncol & 1;
            const float normv = (xy == 0) ? (float)WW : (float)HH;
#pragma unroll
            for (int r = 0; r < 8; ++r) {
                const int q   = q0 + mbase + r;
                const float off = MAXRES * fast_tanhf(acc[r] + bias);
                const float fl  = flow[((size_t)b * 2 + xy) * QQ + q];
                const float ref = (xy == 0)
                    ? (((float)(q & (WW - 1)) + 0.5f) / (float)WW)
                    : (((float)(q >> 7)       + 0.5f) / (float)HH);
                loc_out[(((size_t)b * QQ + q) * (NH * NP) + hp) * 2 + xy] =
                    ref + (fl + off) / normv;
            }
        } else {                                       // attention logits -> LDS
            const float bias = b_attn[ncol];
#pragma unroll
            for (int r = 0; r < 8; ++r)
                ldsAttn[(mbase + r) * 64 + ncol] = acc[r] + bias;
        }
    }
    __syncthreads();

    // Per-(query, head) softmax over the 8 points.
    if (tid < 128) {
        const int m = tid >> 3, head = tid & 7;
        const float* row = &ldsAttn[m * 64 + head * NP];
        float mx = row[0];
#pragma unroll
        for (int p = 1; p < NP; ++p) mx = fmaxf(mx, row[p]);
        float e[NP], s = 0.f;
#pragma unroll
        for (int p = 0; p < NP; ++p) { e[p] = __expf(row[p] - mx); s += e[p]; }
        const float inv = 1.f / s;
        const int q = q0 + m;
#pragma unroll
        for (int p = 0; p < NP; ++p)
            attn_out[(((size_t)b * QQ + q) * NH + head) * NP + p] = e[p] * inv;
    }
}

// ---------------------------------------------------------------------------
// Kernel 2: bilinear gather + attention-weighted aggregation.
// Half-wave per (b, q, head); lane = channel d -> each corner fetch is
// 16 lanes x 16 consecutive floats = one 64B line (L2-resident val buffer).
// ---------------------------------------------------------------------------
__global__ __launch_bounds__(256)
void k_gather(const float* __restrict__ val, const float* __restrict__ loc,
              const float* __restrict__ attn, float* __restrict__ agg)
{
    const size_t t = (size_t)blockIdx.x * 256 + threadIdx.x;
    const int    d    = (int)(t & 15);
    const size_t s    = t >> 4;            // ((b*Q + q)*NH + head)
    const int    head = (int)(s & (NH - 1));
    const size_t bq   = s >> 3;            // b*Q + q
    const int    b    = (int)(bq >> 14);   // Q = 2^14

    const float* vbase = val + ((size_t)b * NH + head) * QQ * DHD;
    const float* lbase = loc  + ((bq * (NH * NP)) + head * NP) * 2;
    const float* abase = attn + (bq * NH + head) * NP;

    float acc = 0.f;
#pragma unroll
    for (int p = 0; p < NP; ++p) {
        const float lx = lbase[p * 2 + 0];
        const float ly = lbase[p * 2 + 1];
        const float aw = abase[p];
        const float x  = lx * (float)WW - 0.5f;
        const float y  = ly * (float)HH - 0.5f;
        const float x0f = floorf(x), y0f = floorf(y);
        const float wx = x - x0f, wy = y - y0f;
        const int x0 = (int)x0f, y0 = (int)y0f;
#pragma unroll
        for (int dy = 0; dy < 2; ++dy) {
#pragma unroll
            for (int dx = 0; dx < 2; ++dx) {
                const int xi = x0 + dx, yi = y0 + dy;
                if (xi >= 0 && xi < WW && yi >= 0 && yi < HH) {
                    const float wgt = (dx ? wx : 1.f - wx) * (dy ? wy : 1.f - wy);
                    acc += aw * wgt * vbase[((size_t)(yi * WW + xi)) * DHD + d];
                }
            }
        }
    }
    agg[bq * CC + head * DHD + d] = acc;
}

// ---------------------------------------------------------------------------
// Kernel 3: out = agg @ W_out + b_out, stored as (B, C, H, W).
// Packed-B tiles 20..27.
// ---------------------------------------------------------------------------
__global__ __launch_bounds__(256)
void k_out(const float* __restrict__ agg, const half_t* __restrict__ wsB,
           const float* __restrict__ b_out, float* __restrict__ out)
{
    __shared__ half_t ldsA[16 * LDAH];

    const int wg  = blockIdx.x;
    const int b   = wg / (QQ / 16);
    const int q0  = (wg % (QQ / 16)) * 16;
    const int tid = threadIdx.x;

    const float* src = agg + ((size_t)b * QQ + q0) * CC;   // row-major, contiguous
    for (int j = tid; j < 2048; j += 256)
        ldsA[(j >> 7) * LDAH + (j & 127)] = (half_t)src[j];
    __syncthreads();

    const int lane = tid & 31;
    const int wave = tid >> 5;
    const int n0   = wave * 16;

    v8f acc = {};
#pragma unroll
    for (int kc = 0; kc < 4; ++kc) {
        v16h af = load_a_frag(ldsA, kc * 32, lane);
        v16h bf = load_b_packed(wsB, 20 + wave, kc, lane);
        acc = __builtin_amdgcn_wmma_f32_16x16x32_f16(
            false, af, false, bf, (short)0, acc, false, false);
    }

    const int ncol  = n0 + (lane & 15);
    const int mbase = (lane >> 4) * 8;
    const float bias = b_out[ncol];
#pragma unroll
    for (int r = 0; r < 8; ++r) {
        const int q = q0 + mbase + r;
        out[((size_t)b * CC + ncol) * QQ + q] = acc[r] + bias;
    }
}

// ---------------------------------------------------------------------------
extern "C" void kernel_launch(void* const* d_in, const int* in_sizes, int n_in,
                              void* d_out, int out_size, void* d_ws, size_t ws_size,
                              hipStream_t stream) {
    (void)in_sizes; (void)n_in; (void)out_size; (void)ws_size;

    const float* nbr    = (const float*)d_in[0];
    const float* ext    = (const float*)d_in[1];
    const float* flow   = (const float*)d_in[2];
    const float* W_off  = (const float*)d_in[3];
    const float* b_off  = (const float*)d_in[4];
    const float* W_attn = (const float*)d_in[5];
    const float* b_attn = (const float*)d_in[6];
    const float* W_val  = (const float*)d_in[7];
    const float* b_val  = (const float*)d_in[8];
    const float* W_out  = (const float*)d_in[9];
    const float* b_out  = (const float*)d_in[10];
    float* out = (float*)d_out;

    // workspace: [packed f16 B fragments | val | loc | attn | agg]
    half_t* wsB   = (half_t*)d_ws;
    float*  fbase = (float*)((char*)d_ws + WS_FLOAT_OFF);
    float* val  = fbase;                                      // B*NH*Q*DH
    float* loc  = val  + (size_t)BB * NH * QQ * DHD;          // B*Q*NH*NP*2
    float* attn = loc  + (size_t)BB * QQ * NH * NP * 2;       // B*Q*NH*NP
    float* agg  = attn + (size_t)BB * QQ * NH * NP;           // B*Q*C

    k_prep<<<PACKB_HALVES / 256, 256, 0, stream>>>(W_val, W_off, W_attn, W_out, wsB);

    k_proj<<<BB * (QQ / 16), 256, 0, stream>>>(
        nbr, ext, flow, b_off, b_attn, b_val, wsB, val, loc, attn);

    k_gather<<<(BB * NH * QQ * DHD) / 256, 256, 0, stream>>>(val, loc, attn, agg);

    k_out<<<BB * (QQ / 16), 256, 0, stream>>>(agg, wsB, b_out, out);
}
